// HATGNN_1340029796805
// MI455X (gfx1250) — matrix operand order repaired and checked by reading, not compiled, not run on hardware
//
#include <hip/hip_runtime.h>
#include <hip/hip_bf16.h>

// ---------------------------------------------------------------------------
// HAT-GNN forward for MI455X (gfx1250, wave32).
//  * q/k projections folded into [NT,F,H] vectors (att vectors absorbed).
//  * V projection done with v_wmma_f32_16x16x32_bf16, weights pre-swizzled
//    into B-fragment order so each lane loads its v16bf with one 32B load.
//  * e_out [E,128] never materialized: e_cls fused into the message kernel.
//  * segment max via order-preserving uint atomics, segment sum via f32 atomics
//    (x_out is 51MB -> L2 resident at 192MB).
//  * float->bf16 via native cast (lowers to v_cvt_pk_bf16_f32, RNE), not
//    manual bit twiddling -- cuts the VALU overhead feeding the WMMAs.
// ---------------------------------------------------------------------------

#define NTYPES 3
#define ETYPES 4
#define FDIM   128
#define HIDDIM 128
#define NHEAD  4
#define DHEAD  32
#define SLOPE  0.2f

typedef __attribute__((ext_vector_type(16))) __bf16 v16bf;
typedef __attribute__((ext_vector_type(8)))  float  v8f;

static __device__ __forceinline__ __bf16 f2bf(float f) {
  return (__bf16)f;  // native RNE convert on gfx1250
}
// order-preserving float<->uint map for atomic max (u=0 acts as -inf)
static __device__ __forceinline__ unsigned mapf(float f) {
  union { float f; unsigned u; } a; a.f = f;
  return (a.u & 0x80000000u) ? ~a.u : (a.u | 0x80000000u);
}
static __device__ __forceinline__ float unmapf(unsigned u) {
  union { unsigned u; float f; } a;
  a.u = (u & 0x80000000u) ? (u ^ 0x80000000u) : ~u;
  return a.f;
}

// ---------------------------------------------------------------------------
// Kernel 0: weight prep.
//  blocks 0..95   : Wv -> bf16 B-fragments  [t][ct][ks] tiles of 32Kx16N
//  blocks 96..107 : folded q/k attn weights -> bf16 B-fragments (cols 0-3 =
//                   Wq.att_dst, cols 4-7 = Wk.att_src, cols 8-15 = 0)
//  block 108      : a_edge[et][h] = dot(be[et] head h, att_edge[h])
// B-fragment packing used here (k order shared with A) (32xK x 16xN, bf16): lane L -> N = L&15,
// K = e + 16*(L>>4) within a 32-K block; lane's 16 elems contiguous (32B).
// ---------------------------------------------------------------------------
__global__ void __launch_bounds__(32)
prep_kernel(const float* __restrict__ Wq, const float* __restrict__ Wk,
            const float* __restrict__ Wv, const float* __restrict__ att_src,
            const float* __restrict__ att_dst, const float* __restrict__ att_edge,
            const float* __restrict__ be,
            __bf16* __restrict__ wvfrag, __bf16* __restrict__ wattn,
            float* __restrict__ aedge) {
  const int bid  = blockIdx.x;
  const int lane = threadIdx.x;
  const int hl = lane >> 4, nn = lane & 15;
  if (bid < 96) {
    const int t = bid >> 5, rem = bid & 31, ct = rem >> 2, ks = rem & 3;
    v16bf frag;
#pragma unroll
    for (int e = 0; e < 16; ++e) {
      const int kg = ks * 32 + e + 16 * hl;
      const int ng = ct * 16 + nn;
      frag[e] = f2bf(Wv[(size_t)t * (FDIM * HIDDIM) + (size_t)kg * HIDDIM + ng]);
    }
    *(v16bf*)(wvfrag + ((size_t)((t * 8 + ct) * 4 + ks)) * 512 + lane * 16) = frag;
  } else if (bid < 108) {
    const int idx = bid - 96, t = idx >> 2, ks = idx & 3;
    v16bf frag;
#pragma unroll
    for (int e = 0; e < 16; ++e) {
      const int kg = ks * 32 + e + 16 * hl;
      float val = 0.f;
      if (nn < 4) {
        const int h = nn;
        const float* wr = Wq + (size_t)t * (FDIM * HIDDIM) + (size_t)kg * HIDDIM + h * DHEAD;
        for (int d = 0; d < DHEAD; ++d) val += wr[d] * att_dst[h * DHEAD + d];
      } else if (nn < 8) {
        const int h = nn - 4;
        const float* wr = Wk + (size_t)t * (FDIM * HIDDIM) + (size_t)kg * HIDDIM + h * DHEAD;
        for (int d = 0; d < DHEAD; ++d) val += wr[d] * att_src[h * DHEAD + d];
      }
      frag[e] = f2bf(val);
    }
    *(v16bf*)(wattn + ((size_t)(t * 4 + ks)) * 512 + lane * 16) = frag;
  } else {
    if (lane < 16) {
      const int et = lane >> 2, h = lane & 3;
      float val = 0.f;
      for (int d = 0; d < DHEAD; ++d)
        val += be[et * HIDDIM + h * DHEAD + d] * att_edge[h * DHEAD + d];
      aedge[et * 4 + h] = val;
    }
  }
}

__global__ void zero_kernel(uint4* __restrict__ dst, size_t n4) {
  size_t i = blockIdx.x * (size_t)blockDim.x + threadIdx.x;
  const size_t stride = (size_t)gridDim.x * blockDim.x;
  const uint4 z = {0u, 0u, 0u, 0u};
  for (; i < n4; i += stride) dst[i] = z;
}

// ---------------------------------------------------------------------------
// Kernel 1: V projection + folded attn scalars. One wave per 16-node tile.
// x tile staged in LDS (row stride 132 floats -> conflict-free swizzled reads).
// 3 accumulator chains (one per node type), per-row select at store.
// Column-tile 8 is the folded attn tile -> a_dst (cols 0-3), a_src (cols 4-7).
// ---------------------------------------------------------------------------
__global__ void __launch_bounds__(32)
proj_kernel(const float* __restrict__ x, const int* __restrict__ node_type, int N,
            const __bf16* __restrict__ wvfrag, const __bf16* __restrict__ wattn,
            float* __restrict__ v, float* __restrict__ a_src,
            float* __restrict__ a_dst) {
  __shared__ float xs[16 * 132];
  const int lane = threadIdx.x;
  const int n0 = blockIdx.x * 16;

  // cooperative coalesced load of the 16x128 tile (rows clamped at tail)
  for (int row = 0; row < 16; ++row) {
    const int rsrc = min(n0 + row, N - 1);
    const float4 val = *(const float4*)(x + (size_t)rsrc * FDIM + lane * 4);
    *(float4*)(xs + row * 132 + lane * 4) = val;
  }
  __syncthreads();  // single-wave WG: lowers to waits (+S_NOP barrier)

  const int hl = lane >> 4, nn = lane & 15, M = nn;

  // A fragments: 16x32 bf16 per k-step. Lane L holds row M=L&15;
  // K = e + 16*hl (contiguous, matching the B-fragment packing).
  v16bf af[4];
#pragma unroll
  for (int ks = 0; ks < 4; ++ks)
#pragma unroll
    for (int e = 0; e < 16; ++e) {
      const int p = e >> 1;
      const int kl = e + 16 * hl;
      af[ks][e] = f2bf(xs[M * 132 + ks * 32 + kl]);
    }

  // node types of this lane's 8 output rows (C/D row M = j + 8*hl)
  int tj[8];
#pragma unroll
  for (int j = 0; j < 8; ++j)
    tj[j] = node_type[min(n0 + j + 8 * hl, N - 1)];

#pragma unroll
  for (int ct = 0; ct < 9; ++ct) {
    v8f acc0 = {}, acc1 = {}, acc2 = {};
    const __bf16* base =
        (ct < 8) ? (wvfrag + (size_t)ct * 4 * 512) : wattn;
    const size_t tstride = (ct < 8) ? (size_t)8 * 4 * 512 : (size_t)4 * 512;
#pragma unroll
    for (int ks = 0; ks < 4; ++ks) {
      const __bf16* bb = base + (size_t)ks * 512 + lane * 16;
      const v16bf b0 = *(const v16bf*)(bb);
      const v16bf b1 = *(const v16bf*)(bb + tstride);
      const v16bf b2 = *(const v16bf*)(bb + 2 * tstride);
      acc0 = __builtin_amdgcn_wmma_f32_16x16x32_bf16(false, af[ks], false, b0,
                                                     (short)0, acc0, false, false);
      acc1 = __builtin_amdgcn_wmma_f32_16x16x32_bf16(false, af[ks], false, b1,
                                                     (short)0, acc1, false, false);
      acc2 = __builtin_amdgcn_wmma_f32_16x16x32_bf16(false, af[ks], false, b2,
                                                     (short)0, acc2, false, false);
    }
#pragma unroll
    for (int j = 0; j < 8; ++j) {
      const int row = n0 + j + 8 * hl;
      if (row < N) {
        const int t = tj[j];
        const float val = (t == 0) ? acc0[j] : ((t == 1) ? acc1[j] : acc2[j]);
        if (ct < 8) {
          v[(size_t)row * HIDDIM + ct * 16 + nn] = val;
        } else if (nn < 4) {
          a_dst[(size_t)row * 4 + nn] = val;
        } else if (nn < 8) {
          a_src[(size_t)row * 4 + (nn - 4)] = val;
        }
      }
    }
  }
}

// ---------------------------------------------------------------------------
// Kernel 2: per-edge attention logits (leaky relu) + segment max via
// order-preserving uint atomic max. One thread per edge, 4 heads.
// ---------------------------------------------------------------------------
__global__ void logits_kernel(const int* __restrict__ edge_index,
                              const int* __restrict__ edge_type, int E,
                              const float* __restrict__ a_src,
                              const float* __restrict__ a_dst,
                              const float* __restrict__ aedge,
                              float* __restrict__ logits, unsigned* __restrict__ m) {
  const int e = blockIdx.x * blockDim.x + threadIdx.x;
  if (e >= E) return;
  const int src = edge_index[e], dst = edge_index[E + e], et = edge_type[e];
  const float4 as = *(const float4*)(a_src + (size_t)src * 4);
  const float4 ad = *(const float4*)(a_dst + (size_t)dst * 4);
  const float4 ae = *(const float4*)(aedge + et * 4);
  float l[4] = {as.x + ad.x + ae.x, as.y + ad.y + ae.y,
                as.z + ad.z + ae.z, as.w + ad.w + ae.w};
#pragma unroll
  for (int h = 0; h < 4; ++h) {
    float lv = l[h];
    lv = (lv >= 0.f) ? lv : SLOPE * lv;
    logits[(size_t)e * 4 + h] = lv;
    atomicMax(m + (size_t)dst * 4 + h, mapf(lv));
  }
}

// Kernel 3: p = exp(logit - max[dst]) in place; denom[dst] += p.
__global__ void expsum_kernel(const int* __restrict__ edge_index, int E,
                              const unsigned* __restrict__ m,
                              float* __restrict__ logits, float* __restrict__ denom) {
  const int e = blockIdx.x * blockDim.x + threadIdx.x;
  if (e >= E) return;
  const int dst = edge_index[E + e];
#pragma unroll
  for (int h = 0; h < 4; ++h) {
    const float mx = unmapf(m[(size_t)dst * 4 + h]);
    const float p = __expf(logits[(size_t)e * 4 + h] - mx);
    logits[(size_t)e * 4 + h] = p;
    atomicAdd(denom + (size_t)dst * 4 + h, p);
  }
}

// ---------------------------------------------------------------------------
// Kernel 4: fused messages. One wave per edge (4 floats/lane):
//   msg = (v[src] + be[et]) * alpha;  x_out[dst] += msg (f32 atomics, L2);
//   e_cls[e] = relu(msg) . Wec[et] + bec[et]  (wave shuffle reduction).
// e_out [E,128] never touches memory.
// ---------------------------------------------------------------------------
__global__ void __launch_bounds__(256)
message_kernel(const int* __restrict__ edge_index, const int* __restrict__ edge_type,
               int E, const float* __restrict__ p, const float* __restrict__ denom,
               const float* __restrict__ v, const float* __restrict__ be,
               const float* __restrict__ Wec, const float* __restrict__ bec,
               float* __restrict__ x_out, float* __restrict__ e_cls) {
  const int lane = threadIdx.x & 31;
  const int wid = (int)((blockIdx.x * blockDim.x + threadIdx.x) >> 5);
  const int nw = (int)((gridDim.x * blockDim.x) >> 5);
  const int h = lane >> 3;       // this lane's 4 cols live in one head block
  const int j0 = lane * 4;
  for (int e = wid; e < E; e += nw) {
    const int src = edge_index[e], dst = edge_index[E + e], et = edge_type[e];
    const float alpha =
        p[(size_t)e * 4 + h] / (denom[(size_t)dst * 4 + h] + 1e-16f);
    const float4 vv = *(const float4*)(v + (size_t)src * HIDDIM + j0);
    const float4 ee = *(const float4*)(be + (size_t)et * HIDDIM + j0);
    float mv[4] = {(vv.x + ee.x) * alpha, (vv.y + ee.y) * alpha,
                   (vv.z + ee.z) * alpha, (vv.w + ee.w) * alpha};
    float part = 0.f;
#pragma unroll
    for (int i = 0; i < 4; ++i) {
      atomicAdd(x_out + (size_t)dst * HIDDIM + j0 + i, mv[i]);
      const float r = mv[i] > 0.f ? mv[i] : 0.f;
      part += r * Wec[(size_t)et * HIDDIM + j0 + i];
    }
#pragma unroll
    for (int off = 16; off > 0; off >>= 1) part += __shfl_down(part, off, 32);
    if (lane == 0) e_cls[e] = part + bec[et];
  }
}

// Kernel 5: x_cls[n] = relu(x_out[n]) . Wn[type[n]] + bn. One wave per node.
__global__ void __launch_bounds__(256)
xcls_kernel(const int* __restrict__ node_type, int N,
            const float* __restrict__ x_out, const float* __restrict__ Wn,
            const float* __restrict__ bn, float* __restrict__ out) {
  const int lane = threadIdx.x & 31;
  const int wid = (int)((blockIdx.x * blockDim.x + threadIdx.x) >> 5);
  const int nw = (int)((gridDim.x * blockDim.x) >> 5);
  const int j0 = lane * 4;
  for (int n = wid; n < N; n += nw) {
    const int t = node_type[n];
    const float4 xo = *(const float4*)(x_out + (size_t)n * HIDDIM + j0);
    float part = fmaxf(xo.x, 0.f) * Wn[t * HIDDIM + j0] +
                 fmaxf(xo.y, 0.f) * Wn[t * HIDDIM + j0 + 1] +
                 fmaxf(xo.z, 0.f) * Wn[t * HIDDIM + j0 + 2] +
                 fmaxf(xo.w, 0.f) * Wn[t * HIDDIM + j0 + 3];
#pragma unroll
    for (int off = 16; off > 0; off >>= 1) part += __shfl_down(part, off, 32);
    if (lane == 0) out[n] = part + bn[t];
  }
}

// ---------------------------------------------------------------------------
extern "C" void kernel_launch(void* const* d_in, const int* in_sizes, int n_in,
                              void* d_out, int out_size, void* d_ws, size_t ws_size,
                              hipStream_t stream) {
  (void)n_in; (void)out_size; (void)ws_size;
  const float* x         = (const float*)d_in[0];
  const int*   edge_index= (const int*)d_in[1];
  const int*   node_type = (const int*)d_in[2];
  const int*   edge_type = (const int*)d_in[3];
  // d_in[4] edge_attr unused: reference forwards zeros into the conv
  const float* Wq        = (const float*)d_in[5];
  const float* Wk        = (const float*)d_in[6];
  const float* Wv        = (const float*)d_in[7];
  const float* att_src   = (const float*)d_in[8];
  const float* att_dst   = (const float*)d_in[9];
  const float* att_edge  = (const float*)d_in[10];
  // d_in[11] We unused (multiplied by zero edge_attr)
  const float* be        = (const float*)d_in[12];
  const float* Wn        = (const float*)d_in[13];
  const float* bn        = (const float*)d_in[14];
  const float* Wec       = (const float*)d_in[15];
  const float* bec       = (const float*)d_in[16];
  const int N = in_sizes[2];
  const int E = in_sizes[3];
  float* out = (float*)d_out;  // [N] x_cls then [E] e_cls

  // workspace layout (all offsets 256B-aligned)
  char* ws = (char*)d_ws;
  auto align256 = [](size_t o) { return (o + 255) & ~(size_t)255; };
  size_t off = 0;
  __bf16* wvfrag = (__bf16*)(ws + off); off = align256(off + 3 * 8 * 4 * 1024);
  __bf16* wattn  = (__bf16*)(ws + off); off = align256(off + 3 * 4 * 1024);
  float* aedge   = (float*)(ws + off);  off = align256(off + ETYPES * NHEAD * 4);
  float* a_srcb  = (float*)(ws + off);  off = align256(off + (size_t)N * 16);
  float* a_dstb  = (float*)(ws + off);  off = align256(off + (size_t)N * 16);
  float* vbuf    = (float*)(ws + off);  off = align256(off + (size_t)N * HIDDIM * 4);
  float* logits  = (float*)(ws + off);  off = align256(off + (size_t)E * 16);
  char*  zbase   = ws + off;            // m, denom, x_out: one contiguous zero span
  unsigned* m    = (unsigned*)(ws + off); off = align256(off + (size_t)N * 16);
  float* denom   = (float*)(ws + off);  off = align256(off + (size_t)N * 16);
  float* x_out   = (float*)(ws + off);  off = align256(off + (size_t)N * HIDDIM * 4);
  const size_t zbytes = (size_t)((ws + off) - zbase);

  prep_kernel<<<109, 32, 0, stream>>>(Wq, Wk, Wv, att_src, att_dst, att_edge,
                                      be, wvfrag, wattn, aedge);
  zero_kernel<<<2048, 256, 0, stream>>>((uint4*)zbase, zbytes / 16);
  proj_kernel<<<(N + 15) / 16, 32, 0, stream>>>(x, node_type, N, wvfrag, wattn,
                                                vbuf, a_srcb, a_dstb);
  logits_kernel<<<(E + 255) / 256, 256, 0, stream>>>(edge_index, edge_type, E,
                                                     a_srcb, a_dstb, aedge,
                                                     logits, m);
  expsum_kernel<<<(E + 255) / 256, 256, 0, stream>>>(edge_index, E, m, logits,
                                                     denom);
  message_kernel<<<4096, 256, 0, stream>>>(edge_index, edge_type, E, logits,
                                           denom, vbuf, be, Wec, bec, x_out,
                                           out + N);
  xcls_kernel<<<4096, 256, 0, stream>>>(node_type, N, x_out, Wn, bn, out);
}
